// MarkonvV_3015067041994
// MI455X (gfx1250) — matrix operand order repaired and checked by reading, not compiled, not run on hardware
//
#include <hip/hip_runtime.h>
#include <math.h>

typedef __attribute__((ext_vector_type(16))) _Float16 v16h;
typedef __attribute__((ext_vector_type(8)))  _Float16 v8h;
typedef __attribute__((ext_vector_type(8)))  float    v8f;

#define KL_   18
#define K2_   64
#define B_    256
#define L_    2000
#define PAD_  4
#define LOUT_ 1990          // L + 2*PAD - KL
#define TM_   64            // positions per workgroup
#define NCHUNK_ 32          // ceil(1990/64)
#define NQ_   9             // K=288 -> 9 chunks of 32
#define FRAG_HALFS_ 512     // 32 lanes * 16 halfs per (dir,ntile,q) fragment
#define BB_   8             // batches per block (persistent loop)

__device__ __forceinline__ float sigmoidf_(float z) {
    return 1.0f / (1.0f + __expf(-z));
}

// Pre-swizzle masked weights into exact per-lane WMMA B-fragment order:
// frag[(((dir*4+n)*9+q)*32 + lane)*16 + h]  where lane holds column n*16+(lane&15),
// K = 32q + (lane>>4)*8 + (h&7) + (h>>3)*16  (lanes0-15 K0..7,16..23; lanes16-31 K8..15,24..31).
__global__ __launch_bounds__(256) void markonv_prep_weights(
    const float* __restrict__ Wf, const float* __restrict__ Wr,
    const float* __restrict__ kwf, const float* __restrict__ kwr,
    _Float16* __restrict__ frag)
{
    int idx = blockIdx.x * blockDim.x + threadIdx.x;
    if (idx >= 2 * 4 * NQ_ * 32 * 16) return;
    int h    = idx & 15;
    int t    = idx >> 4;
    int lane = t & 31;  t >>= 5;
    int q    = t % NQ_; t /= NQ_;
    int n    = t & 3;
    int dir  = t >> 2;

    int K = 32 * q + (lane >> 4) * 8 + (h & 7) + (h >> 3) * 16;   // 0..287
    int j = K >> 4;                          // tap 0..17
    int p = K & 15;                          // (c,d) pair
    int c = p >> 2, d = p & 3;
    int col = n * 16 + (lane & 15);          // output kernel 0..63

    const float* W  = dir ? Wr  : Wf;
    const float* kw = dir ? kwr : kwf;       // [2,1,64] flat
    float m = sigmoidf_((float)j - kw[col]) + sigmoidf_(kw[64 + col] - (float)j) - 1.0f;
    float w = W[((j * 4 + c) * 4 + d) * K2_ + col] * m;
    frag[idx] = (_Float16)w;
}

__global__ __launch_bounds__(256) void markonv_main(
    const float* __restrict__ x,          // [B, L, 4]
    const _Float16* __restrict__ frag,    // pre-swizzled B fragments
    float* __restrict__ out)              // [B, 128, 1990]
{
    // staged padded x windows (fwd/rev) and f16 pair-product rows y[t][16] (stride 24 halfs = 48B)
    __shared__ alignas(16) float    xs[2][84][4];
    __shared__ alignas(16) _Float16 ybuf[2][84][24];

    const int tid   = threadIdx.x;
    const int chunk = blockIdx.x;          // 0..31
    const int bg    = blockIdx.y;          // 0..31 -> batches bg*8 .. bg*8+7
    const int i0    = chunk * TM_;

    const int wave = tid >> 5, lane = tid & 31;
    const int dir  = wave >> 2, n = wave & 3;
    const int hi   = lane >> 4, m = lane & 15;

    // ---- Load B fragments ONCE per block (persistent across 8 batches)
    v16h bfrag[NQ_];
    {
        const v16h* bp = (const v16h*)(frag + (size_t)((dir * 4 + n) * NQ_) * FRAG_HALFS_);
        #pragma unroll
        for (int q = 0; q < NQ_; ++q)
            bfrag[q] = bp[q * 32 + lane];   // 32B per lane, fully coalesced, L2-resident
    }

    const _Float16* yrow = &ybuf[dir][0][hi * 8];   // loop-invariant part of A addressing

    for (int it = 0; it < BB_; ++it) {
        const int b = bg * BB_ + it;

        // separate previous iteration's LDS readers from this iteration's writers
        __syncthreads();

        // ---- Phase 1a: load padded x windows: xp[t] for t = i0 .. i0+81 (both dirs)
        if (tid < 2 * 82) {
            int d0 = tid / 82, tl = tid % 82;
            int pos = d0 ? (2003 - i0 - tl)     // reverse: L-1-(i0+tl-PAD)
                         : (i0 + tl - PAD_);    // forward
            float4 v = make_float4(0.f, 0.f, 0.f, 0.f);
            if (pos >= 0 && pos < L_)
                v = ((const float4*)x)[(size_t)b * L_ + pos];
            *(float4*)xs[d0][tl] = v;
        }
        __syncthreads();

        // ---- Phase 1b: y[p=(c,d), t] = xp[c,t]*xp[d,t+1], t = 0..80, as f16
        if (tid < 2 * 81) {
            int d0 = tid / 81, tl = tid % 81;
            float a0 = xs[d0][tl][0],   a1 = xs[d0][tl][1];
            float a2 = xs[d0][tl][2],   a3 = xs[d0][tl][3];
            float b0 = xs[d0][tl+1][0], b1 = xs[d0][tl+1][1];
            float b2 = xs[d0][tl+1][2], b3 = xs[d0][tl+1][3];
            _Float16* yp = &ybuf[d0][tl][0];
            yp[0]  = (_Float16)(a0*b0); yp[1]  = (_Float16)(a0*b1);
            yp[2]  = (_Float16)(a0*b2); yp[3]  = (_Float16)(a0*b3);
            yp[4]  = (_Float16)(a1*b0); yp[5]  = (_Float16)(a1*b1);
            yp[6]  = (_Float16)(a1*b2); yp[7]  = (_Float16)(a1*b3);
            yp[8]  = (_Float16)(a2*b0); yp[9]  = (_Float16)(a2*b1);
            yp[10] = (_Float16)(a2*b2); yp[11] = (_Float16)(a2*b3);
            yp[12] = (_Float16)(a3*b0); yp[13] = (_Float16)(a3*b1);
            yp[14] = (_Float16)(a3*b2); yp[15] = (_Float16)(a3*b3);
        }
        __syncthreads();

        // ---- Phase 2: 8 waves = (dir 0..1) x (ntile 0..3); 4 M-subtiles x 9 WMMAs each
        float* orow = out + ((size_t)b * 128 + dir * 64 + n * 16 + m) * LOUT_;

        #pragma unroll
        for (int s = 0; s < 4; ++s) {
            v8f acc = {0.f, 0.f, 0.f, 0.f, 0.f, 0.f, 0.f, 0.f};

            // software-pipelined A-fragment loads: issue q+1's loads before q's WMMA
            int tb = s * 16 + m;
            v8h alo = *(const v8h*)(yrow + (tb    ) * 24);   // ds_load_b128
            v8h ahi = *(const v8h*)(yrow + (tb + 1) * 24);   // ds_load_b128
            #pragma unroll
            for (int q = 0; q < NQ_; ++q) {
                v8h clo = alo, chi = ahi;
                if (q + 1 < NQ_) {
                    int t1 = tb + 2 * (q + 1);
                    alo = *(const v8h*)(yrow + (t1    ) * 24);
                    ahi = *(const v8h*)(yrow + (t1 + 1) * 24);
                }
                // A fragment (16-bit 16x32 layout): lo lanes p0..7, hi lanes p8..15,
                // halfs[0..7] from tap j0=2q, halfs[8..15] from tap j0+1
                v16h a = __builtin_shufflevector(clo, chi,
                           0,1,2,3,4,5,6,7,8,9,10,11,12,13,14,15);
                acc = __builtin_amdgcn_wmma_f32_16x16x32_f16(
                          false, a, false, bfrag[q], (short)0, acc, false, false);
            }

            // D layout: lane -> column (m), VGPR r -> position hi*8+r (consecutive in i)
            int ib = i0 + s * 16 + hi * 8;
            if (ib + 7 < LOUT_) {
                // row pitch 1990 floats -> 8-byte aligned: use float2 stores
                *(float2*)(orow + ib)     = make_float2(acc[0], acc[1]);
                *(float2*)(orow + ib + 2) = make_float2(acc[2], acc[3]);
                *(float2*)(orow + ib + 4) = make_float2(acc[4], acc[5]);
                *(float2*)(orow + ib + 6) = make_float2(acc[6], acc[7]);
            } else {
                #pragma unroll
                for (int r = 0; r < 8; ++r)
                    if (ib + r < LOUT_) orow[ib + r] = acc[r];
            }
        }
    }
}

extern "C" void kernel_launch(void* const* d_in, const int* in_sizes, int n_in,
                              void* d_out, int out_size, void* d_ws, size_t ws_size,
                              hipStream_t stream) {
    const float* x   = (const float*)d_in[0];
    const float* Wf  = (const float*)d_in[1];
    const float* Wr  = (const float*)d_in[2];
    const float* kwf = (const float*)d_in[3];
    const float* kwr = (const float*)d_in[4];
    _Float16* frag = (_Float16*)d_ws;   // 2*4*9*32*16 halfs = 73728 B

    markonv_prep_weights<<<dim3(144), dim3(256), 0, stream>>>(Wf, Wr, kwf, kwr, frag);
    markonv_main<<<dim3(NCHUNK_, B_ / BB_), dim3(256), 0, stream>>>(x, frag, (float*)d_out);
}
